// CosineEncoderBlock_61538291417345
// MI455X (gfx1250) — matrix operand order repaired and checked by reading, not compiled, not run on hardware
//
#include <hip/hip_runtime.h>
#include <hip/hip_bf16.h>
#include <math.h>

#define DIM 1024
#define HEADS 16
#define DIM_HEAD 64
#define INNER 1024
#define MLP_DIM 4096
#define NQ 16384
#define NW 1024

typedef __attribute__((ext_vector_type(16))) __bf16 v16bf;
typedef __attribute__((ext_vector_type(8)))  __bf16 bf16x8;
typedef __attribute__((ext_vector_type(8)))  float  v8f;
typedef __attribute__((ext_vector_type(4)))  int    v4i;

union FragBF { v16bf v; bf16x8 h[2]; };

#if defined(__has_builtin)
#if __has_builtin(__builtin_amdgcn_global_load_async_to_lds_b128) && \
    __has_builtin(__builtin_amdgcn_s_wait_asynccnt)
#define USE_ASYNC_LDS 1
#endif
#endif
#ifndef USE_ASYNC_LDS
#define USE_ASYNC_LDS 0
#endif

#if USE_ASYNC_LDS
typedef __attribute__((address_space(1))) v4i v4i_g;   // global int4
typedef __attribute__((address_space(3))) v4i v4i_l;   // LDS int4
__device__ __forceinline__ void async_cp16(const __bf16* g, __bf16* l) {
  __builtin_amdgcn_global_load_async_to_lds_b128((v4i_g*)g, (v4i_l*)l, 0, 0);
}
#endif

__device__ __forceinline__ __bf16 f2bf(float f) {
  unsigned u = __float_as_uint(f);
  u += 0x7FFFu + ((u >> 16) & 1u);           // round-to-nearest-even
  unsigned short s = (unsigned short)(u >> 16);
  return __builtin_bit_cast(__bf16, s);
}
__device__ __forceinline__ float bf2f(__bf16 h) {
  unsigned short s = __builtin_bit_cast(unsigned short, h);
  return __uint_as_float(((unsigned)s) << 16);
}

// ---------------- elementwise fp32 -> bf16 convert (weights) ----------------
__global__ __launch_bounds__(256)
void convert_bf16_kernel(const float* __restrict__ in, __bf16* __restrict__ out, int n) {
  int i = (blockIdx.x * 256 + threadIdx.x) * 4;
  if (i + 3 < n) {
    float4 v = *(const float4*)(in + i);
    out[i + 0] = f2bf(v.x); out[i + 1] = f2bf(v.y);
    out[i + 2] = f2bf(v.z); out[i + 3] = f2bf(v.w);
  }
}

// ---------------- LayerNorm (row=1024) -> bf16 ----------------
__global__ __launch_bounds__(256)
void ln_bf16_kernel(const float* __restrict__ x, const float* __restrict__ w,
                    const float* __restrict__ b, __bf16* __restrict__ out, int) {
  __shared__ float sm[18];
  const int row = blockIdx.x;
  const float4 vv = ((const float4*)(x + (size_t)row * DIM))[threadIdx.x];
  float s  = vv.x + vv.y + vv.z + vv.w;
  float s2 = vv.x * vv.x + vv.y * vv.y + vv.z * vv.z + vv.w * vv.w;
#pragma unroll
  for (int o = 16; o; o >>= 1) { s += __shfl_xor(s, o, 32); s2 += __shfl_xor(s2, o, 32); }
  const int lane = threadIdx.x & 31, wave = threadIdx.x >> 5;
  if (lane == 0) { sm[wave] = s; sm[8 + wave] = s2; }
  __syncthreads();
  if (threadIdx.x == 0) {
    float ts = 0.f, ts2 = 0.f;
    for (int i = 0; i < 8; ++i) { ts += sm[i]; ts2 += sm[8 + i]; }
    float mu  = ts * (1.0f / DIM);
    float var = ts2 * (1.0f / DIM) - mu * mu;
    sm[16] = mu;
    sm[17] = rsqrtf(var + 1e-5f);
  }
  __syncthreads();
  const float mu = sm[16], rs = sm[17];
  const int c = threadIdx.x * 4;
  __bf16* o = out + (size_t)row * DIM + c;
  o[0] = f2bf((vv.x - mu) * rs * w[c + 0] + b[c + 0]);
  o[1] = f2bf((vv.y - mu) * rs * w[c + 1] + b[c + 1]);
  o[2] = f2bf((vv.z - mu) * rs * w[c + 2] + b[c + 2]);
  o[3] = f2bf((vv.w - mu) * rs * w[c + 3] + b[c + 3]);
}

// ---------------- per-head L2 normalize (segments of 64) -> bf16 ----------------
__global__ __launch_bounds__(256)
void l2norm_bf16_kernel(const float* __restrict__ x, __bf16* __restrict__ out, int) {
  const int seg  = blockIdx.x * 8 + (threadIdx.x >> 5);   // one wave32 per 64-chunk
  const int lane = threadIdx.x & 31;
  const float* p = x + (size_t)seg * 64;
  const float a = p[lane], b = p[lane + 32];
  float s = a * a + b * b;
#pragma unroll
  for (int o = 16; o; o >>= 1) s += __shfl_xor(s, o, 32);
  const float inv = 1.0f / fmaxf(sqrtf(s), 1e-12f);
  __bf16* q = out + (size_t)seg * 64;
  q[lane]      = f2bf(a * inv);
  q[lane + 32] = f2bf(b * inv);
}

// ---------------- Mh[h] = khat_h^T (64x1024) @ v_h (1024x64) ----------------
__global__ __launch_bounds__(256)
void ktv_kernel(const __bf16* __restrict__ khat, const float* __restrict__ v,
                float* __restrict__ mh) {
  const int h  = blockIdx.x;
  const int i  = threadIdx.x >> 2;
  const int j0 = (threadIdx.x & 3) * 16;
  float acc[16];
#pragma unroll
  for (int e = 0; e < 16; ++e) acc[e] = 0.f;
  for (int n = 0; n < NW; ++n) {
    const float kv = bf2f(khat[(size_t)n * INNER + h * DIM_HEAD + i]);
    const float* vr = v + (size_t)n * INNER + h * DIM_HEAD + j0;
#pragma unroll
    for (int e = 0; e < 16; ++e) acc[e] += kv * vr[e];
  }
  float* o = mh + (size_t)h * DIM_HEAD * DIM_HEAD + (size_t)i * DIM_HEAD + j0;
#pragma unroll
  for (int e = 0; e < 16; ++e) o[e] = acc[e];
}

// ---------------- P[h*64+i][n] = sum_j Mh[h][i][j] * wo[h*64+j][n]  (bf16 out) ----------------
__global__ __launch_bounds__(256)
void pmat_kernel(const float* __restrict__ mh, const float* __restrict__ wo,
                 __bf16* __restrict__ pb) {
  const size_t idx = (size_t)blockIdx.x * 256 + threadIdx.x;
  const int row = (int)(idx >> 10);  // inner index = h*64+i
  const int n   = (int)(idx & 1023);
  const int h = row >> 6, i = row & 63;
  const float* m = mh + (size_t)h * 4096 + (size_t)i * 64;
  float s = 0.f;
#pragma unroll 8
  for (int j = 0; j < 64; ++j)
    s += m[j] * wo[(size_t)(h * 64 + j) * DIM + n];
  pb[idx] = f2bf(s);
}

// ---------------- WMMA bf16 GEMM: C(MxN) = A(MxK) @ B(KxN), fused epilogues ----------------
enum { EP_NONE = 0, EP_BIAS_RES = 1, EP_GELU_BF16 = 2 };

template <int EPI>
__global__ __launch_bounds__(256)
void gemm_wmma_kernel(const __bf16* __restrict__ A, const __bf16* __restrict__ B,
                      void* __restrict__ Out, const float* __restrict__ bias,
                      const float* __restrict__ res, int M, int N, int K) {
  constexpr int BM = 128, BN = 128, BK = 32;
  constexpr int LDA = BK + 8;   // bf16 elems; 80B row stride (16B aligned)
  constexpr int LDB = BN + 8;   // 272B row stride (16B aligned)

  const int tid  = threadIdx.x;
  const int lane = tid & 31;
  const int wave = tid >> 5;
  const int wm   = (wave & 1) * 64;   // wave 64-row slice
  const int wn   = (wave >> 1) * 32;  // wave 32-col slice
  const int r    = lane & 15;
  const int half = lane >> 4;

  const int bm = blockIdx.y * BM;
  const int bn = blockIdx.x * BN;

  const __bf16* Ag = A + (size_t)bm * K;
  const __bf16* Bg = B + bn;

  const int arow = tid >> 2, acol = (tid & 3) * 8;   // A tile: 128x32, 2 chunks/thread
  const int brow = tid >> 4, bcol = (tid & 15) * 8;  // B tile: 32x128, 2 chunks/thread

  // per-thread global pointers, advanced by constant strides (no in-loop mul)
  const __bf16* aP0 = Ag + (size_t)arow * K + acol;
  const __bf16* aP1 = aP0 + (size_t)64 * K;
  const __bf16* bP0 = Bg + (size_t)brow * N + bcol;
  const __bf16* bP1 = bP0 + (size_t)16 * N;
  const size_t bstep = (size_t)BK * N;

  v8f acc[4][2] = {};

  // frag loads + 8 WMMAs for one BK=32 step, from given LDS tile bases
  auto compute = [&](const __bf16* Ab, const __bf16* Bb) {
    FragBF a[4], b[2];
#pragma unroll
    for (int im = 0; im < 4; ++im) {
      // 16-bit A 16x32 layout: lane(M-row)=r, half selects K groups {0..7,16..23}/{8..15,24..31}
      const __bf16* ap = &Ab[(size_t)(wm + im * 16 + r) * LDA + half * 8];
      a[im].h[0] = *(const bf16x8*)ap;
      a[im].h[1] = *(const bf16x8*)(ap + 16);
    }
#pragma unroll
    for (int in_ = 0; in_ < 2; ++in_) {
      // 16-bit B 32x16 layout: lane(N-col)=r, half selects K groups {0..7,16..23}/{8..15,24..31}
      const __bf16* bp = &Bb[wn + in_ * 16 + r];
#pragma unroll
      for (int s = 0; s < 16; ++s)
        b[in_].v[s] = bp[(size_t)((s < 8 ? half * 8 + s : 16 + half * 8 + (s - 8))) * LDB];
    }
#pragma unroll
    for (int im = 0; im < 4; ++im)
#pragma unroll
      for (int in_ = 0; in_ < 2; ++in_)
        acc[im][in_] = __builtin_amdgcn_wmma_f32_16x16x32_bf16(
            false, a[im].v, false, b[in_].v, (short)0, acc[im][in_], false, false);
  };

#if USE_ASYNC_LDS
  // ---- double-buffered LDS, gfx1250 async global->LDS copies (ASYNCcnt) ----
  // K/BK is even for every GEMM here, so unroll 2x with literal buffer indices:
  // no runtime buffer selects, LDS bases fold into ds_load offsets.
  __shared__ __bf16 As[2][BM * LDA];
  __shared__ __bf16 Bs[2][BK * LDB];
  __bf16* const lA0_0 = &As[0][arow * LDA + acol];
  __bf16* const lA1_0 = &As[0][(arow + 64) * LDA + acol];
  __bf16* const lB0_0 = &Bs[0][brow * LDB + bcol];
  __bf16* const lB1_0 = &Bs[0][(brow + 16) * LDB + bcol];
  __bf16* const lA0_1 = &As[1][arow * LDA + acol];
  __bf16* const lA1_1 = &As[1][(arow + 64) * LDA + acol];
  __bf16* const lB0_1 = &Bs[1][brow * LDB + bcol];
  __bf16* const lB1_1 = &Bs[1][(brow + 16) * LDB + bcol];

  // prologue: tile 0 -> buffer 0
  async_cp16(aP0, lA0_0); async_cp16(aP1, lA1_0);
  async_cp16(bP0, lB0_0); async_cp16(bP1, lB1_0);

  const int nk = K / BK;                    // even (32 or 128)
  for (int kt = 0; kt < nk; kt += 2) {
    // ---- tile kt from buf0; prefetch tile kt+1 -> buf1 (always exists) ----
    aP0 += BK; aP1 += BK; bP0 += bstep; bP1 += bstep;
    async_cp16(aP0, lA0_1); async_cp16(aP1, lA1_1);
    async_cp16(bP0, lB0_1); async_cp16(bP1, lB1_1);
    __builtin_amdgcn_s_wait_asynccnt(4);    // in-order: oldest 4 (buf0 tile) done
    __syncthreads();
    compute(As[0], Bs[0]);
    __syncthreads();

    // ---- tile kt+1 from buf1; prefetch tile kt+2 -> buf0 if it exists ----
    if (kt + 2 < nk) {
      aP0 += BK; aP1 += BK; bP0 += bstep; bP1 += bstep;
      async_cp16(aP0, lA0_0); async_cp16(aP1, lA1_0);
      async_cp16(bP0, lB0_0); async_cp16(bP1, lB1_0);
      __builtin_amdgcn_s_wait_asynccnt(4);
    } else {
      __builtin_amdgcn_s_wait_asynccnt(0);
    }
    __syncthreads();
    compute(As[1], Bs[1]);
    __syncthreads();
  }
#else
  // ---- fallback: register-staged software pipeline, single LDS buffer ----
  __shared__ __bf16 As[BM * LDA];
  __shared__ __bf16 Bs[BK * LDB];
  bf16x8 ra0 = *(const bf16x8*)aP0, ra1 = *(const bf16x8*)aP1;
  bf16x8 rb0 = *(const bf16x8*)bP0, rb1 = *(const bf16x8*)bP1;

  for (int k0 = 0; k0 < K; k0 += BK) {
    *(bf16x8*)&As[arow * LDA + acol]        = ra0;
    *(bf16x8*)&As[(arow + 64) * LDA + acol] = ra1;
    *(bf16x8*)&Bs[brow * LDB + bcol]        = rb0;
    *(bf16x8*)&Bs[(brow + 16) * LDB + bcol] = rb1;
    __syncthreads();
    if (k0 + BK < K) {   // next-tile loads overlap the WMMA block below
      aP0 += BK; aP1 += BK; bP0 += bstep; bP1 += bstep;
      ra0 = *(const bf16x8*)aP0; ra1 = *(const bf16x8*)aP1;
      rb0 = *(const bf16x8*)bP0; rb1 = *(const bf16x8*)bP1;
    }
    compute(As, Bs);
    __syncthreads();
  }
#endif

  // epilogue: C 16x16 f32 layout -> VGPR e holds M = e + 8*half, N = lane&15
#pragma unroll
  for (int im = 0; im < 4; ++im) {
#pragma unroll
    for (int in_ = 0; in_ < 2; ++in_) {
#pragma unroll
      for (int e = 0; e < 8; ++e) {
        const int gm = bm + wm + im * 16 + half * 8 + e;
        const int gn = bn + wn + in_ * 16 + r;
        const size_t off = (size_t)gm * N + gn;
        const float x = acc[im][in_][e];
        if (EPI == EP_NONE) {
          ((float*)Out)[off] = x;
        } else if (EPI == EP_BIAS_RES) {
          ((float*)Out)[off] = x + bias[gn] + res[off];
        } else {  // exact GELU -> bf16
          const float y = x + bias[gn];
          ((__bf16*)Out)[off] = f2bf(0.5f * y * (1.0f + erff(y * 0.70710678118654752f)));
        }
      }
    }
  }
}

extern "C" void kernel_launch(void* const* d_in, const int* in_sizes, int n_in,
                              void* d_out, int out_size, void* d_ws, size_t ws_size,
                              hipStream_t stream) {
  (void)in_sizes; (void)n_in; (void)out_size; (void)ws_size;
  const float* queries    = (const float*)d_in[0];
  const float* prototypes = (const float*)d_in[1];
  const float* ln1_w = (const float*)d_in[2];
  const float* ln1_b = (const float*)d_in[3];
  const float* wq = (const float*)d_in[4];
  const float* wk = (const float*)d_in[5];
  const float* wv = (const float*)d_in[6];
  const float* wo = (const float*)d_in[7];
  const float* bo = (const float*)d_in[8];
  const float* ln2_w = (const float*)d_in[9];
  const float* ln2_b = (const float*)d_in[10];
  const float* w1 = (const float*)d_in[11];
  const float* b1 = (const float*)d_in[12];
  const float* w2 = (const float*)d_in[13];
  const float* b2 = (const float*)d_in[14];
  float* out = (float*)d_out;

  char* ws = (char*)d_ws;
  size_t off = 0;
  auto carve = [&](size_t bytes) -> char* {
    char* p = ws + off;
    off += (bytes + 255) & ~(size_t)255;
    return p;
  };
  __bf16* qn   = (__bf16*)carve((size_t)NQ * DIM * 2);
  __bf16* pn   = (__bf16*)carve((size_t)NW * DIM * 2);
  __bf16* wqb  = (__bf16*)carve((size_t)DIM * INNER * 2);
  __bf16* wkb  = (__bf16*)carve((size_t)DIM * INNER * 2);
  __bf16* wvb  = (__bf16*)carve((size_t)DIM * INNER * 2);
  __bf16* w1b  = (__bf16*)carve((size_t)DIM * MLP_DIM * 2);
  __bf16* w2b  = (__bf16*)carve((size_t)MLP_DIM * DIM * 2);
  float*  qf   = (float*)carve((size_t)NQ * INNER * 4);   // later reused as attn_out
  __bf16* qhat = (__bf16*)carve((size_t)NQ * INNER * 2);
  float*  kf   = (float*)carve((size_t)NW * INNER * 4);
  float*  vf   = (float*)carve((size_t)NW * INNER * 4);
  __bf16* khat = (__bf16*)carve((size_t)NW * INNER * 2);
  float*  mh   = (float*)carve((size_t)HEADS * DIM_HEAD * DIM_HEAD * 4);
  __bf16* pb   = (__bf16*)carve((size_t)INNER * DIM * 2);
  __bf16* hb   = (__bf16*)carve((size_t)NQ * DIM * 2);
  __bf16* tb   = (__bf16*)carve((size_t)NQ * MLP_DIM * 2);

  // 1) weights -> bf16
  convert_bf16_kernel<<<(DIM * INNER) / 1024, 256, 0, stream>>>(wq, wqb, DIM * INNER);
  convert_bf16_kernel<<<(DIM * INNER) / 1024, 256, 0, stream>>>(wk, wkb, DIM * INNER);
  convert_bf16_kernel<<<(DIM * INNER) / 1024, 256, 0, stream>>>(wv, wvb, DIM * INNER);
  convert_bf16_kernel<<<(DIM * MLP_DIM) / 1024, 256, 0, stream>>>(w1, w1b, DIM * MLP_DIM);
  convert_bf16_kernel<<<(MLP_DIM * DIM) / 1024, 256, 0, stream>>>(w2, w2b, MLP_DIM * DIM);

  // 2) LN1 on queries & prototypes
  ln_bf16_kernel<<<NQ, 256, 0, stream>>>(queries, ln1_w, ln1_b, qn, NQ);
  ln_bf16_kernel<<<NW, 256, 0, stream>>>(prototypes, ln1_w, ln1_b, pn, NW);

  // 3) projections
  dim3 gq(INNER / 128, NQ / 128);
  gemm_wmma_kernel<EP_NONE><<<gq, 256, 0, stream>>>(qn, wqb, qf, nullptr, nullptr, NQ, INNER, DIM);
  dim3 gp(INNER / 128, NW / 128);
  gemm_wmma_kernel<EP_NONE><<<gp, 256, 0, stream>>>(pn, wkb, kf, nullptr, nullptr, NW, INNER, DIM);
  gemm_wmma_kernel<EP_NONE><<<gp, 256, 0, stream>>>(pn, wvb, vf, nullptr, nullptr, NW, INNER, DIM);

  // 4) per-head L2 norm
  l2norm_bf16_kernel<<<(NQ * HEADS) / 8, 256, 0, stream>>>(qf, qhat, NQ * HEADS);
  l2norm_bf16_kernel<<<(NW * HEADS) / 8, 256, 0, stream>>>(kf, khat, NW * HEADS);

  // 5) Mh = khat^T v per head; P = stack_h(Mh @ Wo_h)   (attention folded into 1024x1024 matrix)
  ktv_kernel<<<HEADS, 256, 0, stream>>>(khat, vf, mh);
  pmat_kernel<<<(INNER * DIM) / 256, 256, 0, stream>>>(mh, wo, pb);

  // 6) attn_out = qhat @ P + bo + queries   (reuse qf buffer; qf no longer needed)
  float* attn_out = qf;
  dim3 ga(DIM / 128, NQ / 128);
  gemm_wmma_kernel<EP_BIAS_RES><<<ga, 256, 0, stream>>>(qhat, pb, attn_out, bo, queries, NQ, DIM, INNER);

  // 7) LN2
  ln_bf16_kernel<<<NQ, 256, 0, stream>>>(attn_out, ln2_w, ln2_b, hb, NQ);

  // 8) MLP: t = GELU(h @ w1 + b1);  out = t @ w2 + b2 + attn_out
  dim3 g1(MLP_DIM / 128, NQ / 128);
  gemm_wmma_kernel<EP_GELU_BF16><<<g1, 256, 0, stream>>>(hb, w1b, tb, b1, nullptr, NQ, MLP_DIM, DIM);
  dim3 g2(DIM / 128, NQ / 128);
  gemm_wmma_kernel<EP_BIAS_RES><<<g2, 256, 0, stream>>>(tb, w2b, out, b2, attn_out, NQ, DIM, MLP_DIM);
}
